// EncoderGRUODE_41394894798854
// MI455X (gfx1250) — matrix operand order repaired and checked by reading, not compiled, not run on hardware
//
#include <hip/hip_runtime.h>

// ---------------------------------------------------------------------------
// GRU-ODE encoder for MI455X (gfx1250, wave32, WMMA).
// B=256, T=512, D_IN=64, H=128.
// Each workgroup (256 thr = 8 wave32) owns 16 batch rows for all 512 steps.
// Each wave owns a 16-column slice. All GEMMs via v_wmma_f32_16x16x32_f16
// with f16 weights resident in LDS and f32 accumulation.
// Activations use gfx1250 hardware transcendentals (v_tanh_f32 / v_exp_f32).
// ---------------------------------------------------------------------------

typedef __attribute__((ext_vector_type(16))) _Float16 v16h;
typedef __attribute__((ext_vector_type(8)))  float    v8f;

#define B_   256
#define T_   512
#define DIN  64
#define H_   128
#define HS   136   // padded LDS row stride (halves) for K=128 matrices
#define DS   72    // padded LDS row stride (halves) for K=64 matrices

// --- LDS layout (in halves) -------------------------------------------------
#define OFF_WN    0
#define OFF_WHH   (OFF_WN   + 128*HS)       // W_node  [128 x 128]
#define OFF_WOUT  (OFF_WHH  + 384*HS)       // W_hh    [384 x 128]
#define OFF_WIH   (OFF_WOUT + 64*HS)        // W_out   [ 64 x 128]
#define OFF_H16   (OFF_WIH  + 384*DS)       // W_ih    [384 x  64]
#define OFF_TA    (OFF_H16  + 16*HS)        // h (f16 shadow)
#define OFF_TB    (OFF_TA   + 16*HS)        // RK4 stage ping
#define OFF_HODE  (OFF_TB   + 16*HS)        // RK4 stage pong
#define OFF_INP   (OFF_HODE + 16*HS)        // h_ode f16
#define OFF_POUT  (OFF_INP  + 16*DS)        // x_t f16
#define HALVES_TOTAL (OFF_POUT + 16*DS)     // prev_out f16
#define NFLOATS   (128 + 384 + 384 + 64 + 512)   // b_node,b_ih,b_hh,b_out,tp
#define SMEM_BYTES (HALVES_TOTAL*2 + NFLOATS*4 + 512 /*mask*/)

// --- fast activations: gfx1250 hardware transcendentals ---------------------
__device__ __forceinline__ float fast_tanh(float v) {
#if __has_builtin(__builtin_amdgcn_tanhf)
    return __builtin_amdgcn_tanhf(v);          // v_tanh_f32 (TRANS op)
#else
    return tanhf(v);
#endif
}
__device__ __forceinline__ float fast_sigmoid(float v) {
#if __has_builtin(__builtin_amdgcn_exp2f) && __has_builtin(__builtin_amdgcn_rcpf)
    // sigmoid(x) = 1 / (1 + 2^(-x*log2(e)))  -> v_exp_f32 + v_rcp_f32
    return __builtin_amdgcn_rcpf(
        1.0f + __builtin_amdgcn_exp2f(v * -1.44269504088896340736f));
#else
    return 1.0f / (1.0f + __expf(-v));
#endif
}

// --- WMMA fragment loaders (contiguous k order, same in A and B) -------------------------
// A matrix 16x32 f16: lane m(0..15)/m+16 -> row m; halves k0..k0+15
// for lanes 0-15, k0+16..k0+31 for lanes 16-31 (contiguous).
__device__ __forceinline__ v16h load_fragA(const _Float16* A, int stride, int k0, int lane) {
    const _Float16* p = A + (lane & 15) * stride + k0 + ((lane >> 4) << 4);
    union { struct { uint4 a, b; } q; v16h v; } u;
    u.q.a = *(const uint4*)(p);
    u.q.b = *(const uint4*)(p + 8);
    return u.v;
}
// B matrix 32x16 f16 (B = W^T): lane n(0..15)/n+16 -> column n = W row n;
// lanes 0-15 hold K=k0..k0+15, lanes 16-31 hold K=k0+16..k0+31 (contiguous).
__device__ __forceinline__ v16h load_fragB(const _Float16* W, int stride, int n0, int k0, int lane) {
    const _Float16* p = W + (n0 + (lane & 15)) * stride + k0 + ((lane >> 4) << 4);
    union { struct { uint4 a, b; } q; v16h v; } u;
    u.q.a = *(const uint4*)(p);
    u.q.b = *(const uint4*)(p + 8);
    return u.v;
}

__device__ __forceinline__ v8f wmma32(v16h a, v16h b, v8f c) {
    return __builtin_amdgcn_wmma_f32_16x16x32_f16(false, a, false, b, (short)0, c, false, false);
}

// D = A[16xK] * W^T[Kx16] (tile at output columns n0..n0+15), K = 32*kblocks
__device__ __forceinline__ v8f gemm_tile(const _Float16* A, int as,
                                         const _Float16* W, int ws,
                                         int n0, int kblocks, int lane) {
    v8f acc = {};
    #pragma unroll
    for (int kb = 0; kb < kblocks; ++kb) {
        v16h a = load_fragA(A, as, kb * 32, lane);
        v16h b = load_fragB(W, ws, n0, kb * 32, lane);
        acc = wmma32(a, b, acc);
    }
    return acc;
}

// three output tiles (gate r/z/n at n0, 128+n0, 256+n0) sharing A fragments
__device__ __forceinline__ void gemm3(const _Float16* A, int as,
                                      const _Float16* W, int ws,
                                      int n0, int kblocks, int lane,
                                      v8f& d0, v8f& d1, v8f& d2) {
    #pragma unroll
    for (int kb = 0; kb < kblocks; ++kb) {
        v16h a  = load_fragA(A, as, kb * 32, lane);
        v16h b0 = load_fragB(W, ws, n0,       kb * 32, lane);
        d0 = wmma32(a, b0, d0);
        v16h b1 = load_fragB(W, ws, 128 + n0, kb * 32, lane);
        d1 = wmma32(a, b1, d1);
        v16h b2 = load_fragB(W, ws, 256 + n0, kb * 32, lane);
        d2 = wmma32(a, b2, d2);
    }
}

// C/D tile: VGPR i holds row (i + 8*(lane>=16)), column n0 + (lane&15)
__device__ __forceinline__ void store_tile_f16(_Float16* dst, int stride, int n0,
                                               v8f v, int lane) {
    int col = n0 + (lane & 15);
    int rb  = (lane >> 4) * 8;
    #pragma unroll
    for (int i = 0; i < 8; ++i) dst[(rb + i) * stride + col] = (_Float16)v[i];
}

__device__ __forceinline__ v8f act_tanh8(v8f a, float b) {
    v8f r;
    #pragma unroll
    for (int i = 0; i < 8; ++i) r[i] = fast_tanh(a[i] + b);
    return r;
}
__device__ __forceinline__ v8f fma8(v8f h, float s, v8f k) {   // h + s*k
    v8f r;
    #pragma unroll
    for (int i = 0; i < 8; ++i) r[i] = __builtin_fmaf(s, k[i], h[i]);
    return r;
}

__global__ void __launch_bounds__(256, 1)
EncoderGRUODE_41394894798854_kernel(const float* __restrict__ x,
                                    const float* __restrict__ tp,
                                    const unsigned char* __restrict__ mask,
                                    const float* __restrict__ Wih,
                                    const float* __restrict__ Whh,
                                    const float* __restrict__ bih,
                                    const float* __restrict__ bhh,
                                    const float* __restrict__ Wn,
                                    const float* __restrict__ bn,
                                    const float* __restrict__ Wout,
                                    const float* __restrict__ bo,
                                    float* __restrict__ out) {
    extern __shared__ char smemraw[];
    _Float16* sBase = (_Float16*)smemraw;
    _Float16* sWn   = sBase + OFF_WN;
    _Float16* sWhh  = sBase + OFF_WHH;
    _Float16* sWout = sBase + OFF_WOUT;
    _Float16* sWih  = sBase + OFF_WIH;
    _Float16* sH16  = sBase + OFF_H16;
    _Float16* sTA   = sBase + OFF_TA;
    _Float16* sTB   = sBase + OFF_TB;
    _Float16* sHODE = sBase + OFF_HODE;
    _Float16* sInp  = sBase + OFF_INP;
    _Float16* sPout = sBase + OFF_POUT;
    float* sF   = (float*)(smemraw + HALVES_TOTAL * 2);
    float* sBn  = sF;
    float* sBih = sBn + 128;
    float* sBhh = sBih + 384;
    float* sBo  = sBhh + 384;
    float* sTp  = sBo + 64;
    unsigned char* sMask = (unsigned char*)(sF + NFLOATS);

    const int tid  = threadIdx.x;
    const int lane = tid & 31;
    const int wave = tid >> 5;
    const int r0   = blockIdx.x * 16;   // batch-row tile base

    // ---- one-time: stage weights (f32 global -> f16 LDS), biases, tp, mask
    for (int i = tid; i < 128 * H_;  i += 256) sWn  [(i >> 7) * HS + (i & 127)] = (_Float16)Wn[i];
    for (int i = tid; i < 384 * H_;  i += 256) sWhh [(i >> 7) * HS + (i & 127)] = (_Float16)Whh[i];
    for (int i = tid; i < 384 * DIN; i += 256) sWih [(i >> 6) * DS + (i &  63)] = (_Float16)Wih[i];
    for (int i = tid; i <  64 * H_;  i += 256) sWout[(i >> 7) * HS + (i & 127)] = (_Float16)Wout[i];
    for (int i = tid; i < 128; i += 256) sBn[i]  = bn[i];
    for (int i = tid; i < 384; i += 256) sBih[i] = bih[i];
    for (int i = tid; i < 384; i += 256) sBhh[i] = bhh[i];
    for (int i = tid; i <  64; i += 256) sBo[i]  = bo[i];
    for (int i = tid; i < T_;  i += 256) sTp[i]  = tp[i];       // row 0 of tp
    for (int i = tid; i < T_;  i += 256) sMask[i] = mask[i];
    for (int i = tid; i < 16 * HS; i += 256) sH16[i] = (_Float16)0.0f;  // h0 = 0
    for (int i = tid; i < 16 * DS; i += 256) {                  // prev_out(h0) = b_out
        int c = i % DS;
        sPout[i] = (c < DIN) ? (_Float16)bo[c] : (_Float16)0.0f;
    }
    __syncthreads();

    const int n0   = wave * 16;          // this wave's H-column slice
    const int colL = lane & 15;
    const float bnv = sBn[n0 + colL];
    v8f hreg = {};                       // f32 master state, this wave's slice

    #pragma unroll 1
    for (int t = 0; t < T_; ++t) {
        const float dt = (t == 0) ? 0.01f : (sTp[t] - sTp[t - 1]);

        // prefetch next sampled step's x rows (16 cachelines) into caches
        if (tid < 16 && (t + 1) < T_ && sMask[t + 1]) {
            __builtin_prefetch(&x[((size_t)(r0 + tid) * T_ + (t + 1)) * DIN], 0, 0);
        }

        // ---- RK4: dh/dt = tanh(h @ Wn^T + bn), ping-pong stage buffers ----
        v8f k1 = act_tanh8(gemm_tile(sH16, HS, sWn, HS, n0, 4, lane), bnv);
        v8f ks = k1;
        store_tile_f16(sTA, HS, n0, fma8(hreg, 0.5f * dt, k1), lane);
        __syncthreads();
        v8f k2 = act_tanh8(gemm_tile(sTA, HS, sWn, HS, n0, 4, lane), bnv);
        #pragma unroll
        for (int i = 0; i < 8; ++i) ks[i] += 2.0f * k2[i];
        store_tile_f16(sTB, HS, n0, fma8(hreg, 0.5f * dt, k2), lane);
        __syncthreads();
        v8f k3 = act_tanh8(gemm_tile(sTB, HS, sWn, HS, n0, 4, lane), bnv);
        #pragma unroll
        for (int i = 0; i < 8; ++i) ks[i] += 2.0f * k3[i];
        store_tile_f16(sTA, HS, n0, fma8(hreg, dt, k3), lane);
        __syncthreads();
        v8f k4 = act_tanh8(gemm_tile(sTA, HS, sWn, HS, n0, 4, lane), bnv);
        #pragma unroll
        for (int i = 0; i < 8; ++i) ks[i] += k4[i];
        v8f hode = fma8(hreg, dt / 6.0f, ks);
        store_tile_f16(sHODE, HS, n0, hode, lane);

        // ---- GRU input: x_t if sampled, else cached prev_out ----
        const bool use_x = (sMask[t] != 0);
        if (use_x) {
            for (int i = tid; i < 16 * DIN; i += 256) {
                int r = i >> 6, c = i & 63;
                sInp[r * DS + c] = (_Float16)x[((size_t)(r0 + r) * T_ + t) * DIN + c];
            }
        }
        __syncthreads();

        // ---- GRU cell: gi = inp@Wih^T + bih ; gh = h_ode@Whh^T + bhh ----
        const _Float16* Ain = use_x ? sInp : sPout;
        v8f gir = {}, giz = {}, gin = {};
        gemm3(Ain, DS, sWih, DS, n0, 2, lane, gir, giz, gin);
        v8f ghr = {}, ghz = {}, ghn = {};
        gemm3(sHODE, HS, sWhh, HS, n0, 4, lane, ghr, ghz, ghn);
        const float bir = sBih[n0 + colL],       bhr = sBhh[n0 + colL];
        const float biz = sBih[128 + n0 + colL], bhz = sBhh[128 + n0 + colL];
        const float bin = sBih[256 + n0 + colL], bhn = sBhh[256 + n0 + colL];
        #pragma unroll
        for (int i = 0; i < 8; ++i) {
            float r = fast_sigmoid((gir[i] + bir) + (ghr[i] + bhr));
            float z = fast_sigmoid((giz[i] + biz) + (ghz[i] + bhz));
            float n = fast_tanh((gin[i] + bin) + r * (ghn[i] + bhn));
            hreg[i] = (1.0f - z) * n + z * hode[i];
        }
        store_tile_f16(sH16, HS, n0, hreg, lane);
        __syncthreads();

        // ---- out_t = h_new @ Wout^T + bo : written to global AND cached as
        //      next step's prev_out (waves 0..3 cover the 64 output columns)
        if (wave < 4) {
            v8f o = gemm_tile(sH16, HS, sWout, HS, n0, 4, lane);
            const float bov = sBo[n0 + colL];
            const int col = n0 + colL;
            const int rb  = (lane >> 4) * 8;
            #pragma unroll
            for (int i = 0; i < 8; ++i) {
                float val = o[i] + bov;
                out[((size_t)(r0 + rb + i) * T_ + t) * DIN + col] = val;
                sPout[(rb + i) * DS + col] = (_Float16)val;
            }
        }
        // prev_out consumption at t+1 is ordered by that step's syncthreads
    }
}

extern "C" void kernel_launch(void* const* d_in, const int* in_sizes, int n_in,
                              void* d_out, int out_size, void* d_ws, size_t ws_size,
                              hipStream_t stream) {
    const float*         x    = (const float*)d_in[0];
    const float*         tp   = (const float*)d_in[1];
    const unsigned char* msk  = (const unsigned char*)d_in[2];   // bool array
    const float*         Wih  = (const float*)d_in[3];
    const float*         Whh  = (const float*)d_in[4];
    const float*         bih  = (const float*)d_in[5];
    const float*         bhh  = (const float*)d_in[6];
    const float*         Wn   = (const float*)d_in[7];
    const float*         bn   = (const float*)d_in[8];
    const float*         Wout = (const float*)d_in[9];
    const float*         bo   = (const float*)d_in[10];
    float*               outp = (float*)d_out;

    // opt in to >64KB dynamic LDS (gfx1250 WGP supports up to 320KB)
    hipFuncSetAttribute((const void*)EncoderGRUODE_41394894798854_kernel,
                        hipFuncAttributeMaxDynamicSharedMemorySize, SMEM_BYTES);

    EncoderGRUODE_41394894798854_kernel<<<dim3(B_ / 16), dim3(256), SMEM_BYTES, stream>>>(
        x, tp, msk, Wih, Whh, bih, bhh, Wn, bn, Wout, bo, outp);
}
